// GHN_24816321036468
// MI455X (gfx1250) — matrix operand order repaired and checked by reading, not compiled, not run on hardware
//
#include <hip/hip_runtime.h>
#include <hip/hip_bf16.h>
#include <hip/hip_fp16.h>

typedef __attribute__((ext_vector_type(16))) _Float16     v16h;
typedef __attribute__((ext_vector_type(8)))  _Float16     v8h;
typedef __attribute__((ext_vector_type(8)))  float        v8f;
typedef __attribute__((ext_vector_type(4)))  unsigned int u32x4;
typedef __attribute__((ext_vector_type(8)))  int          i32x8;
typedef __attribute__((ext_vector_type(4)))  int          i32x4;

#define HID   32
#define NN    1025          // 1 + NC + NL + NB
#define NPAD  1056          // padded to multiple of 32 for WMMA K
#define NC    512
#define NL    256
#define NB    256
#define FD    96            // 3*HID
#define KC    36864         // 64*64*3*3 (valid conv weights per node)
#define KL    4096          // 64*64
#define KB    64
#define PC    147456        // 128*128*3*3 padded conv row
#define PL    16384         // 128*128
#define PB    128

// ---------------------------------------------------------------------------
// WMMA 16x16 tile: C(16x16,f32) += A(16x32) * B(32x16) over K, f16 inputs.
// Fragment layouts used by this code (contiguous k order, not the ISA table order):
//   A (16-bit, 16x32): lanes 0-15 -> M=lane, halves 0..15 = K 0..15;
//                      lanes 16-31 -> M=lane-16, K 16..31.
//   B (32x16): lanes 0-15 hold K 0..15 (N=lane), lanes 16-31 hold K 16..31.
//   C/D (f32): VGPR v, lanes 0-15 -> M=v, N=lane; lanes 16-31 -> M=v+8.
// ---------------------------------------------------------------------------
__device__ __forceinline__ v8f wmma_step(v16h a, v16h b, v8f c) {
#if defined(__gfx1250__)
  return __builtin_amdgcn_wmma_f32_16x16x32_f16(
      /*neg_a=*/false, a, /*neg_b=*/false, b,
      /*c_mod=*/(short)0, c, /*reuse_a=*/false, /*reuse_b=*/false);
#else
  for (int t = 0; t < 8; ++t) c[t] += (float)a[t] * (float)b[t];  // host pass
  return c;
#endif
}

template <typename TA>
__device__ __forceinline__ v8f wmma_tile(const TA* __restrict__ A, int lda,
                                         const _Float16* __restrict__ B, int ldb,
                                         int K, int m0, int n0) {
  const int lane = threadIdx.x & 31;
  const int hs   = lane >> 4;        // half-wave select
  const int lr   = lane & 15;
  const int akb  = hs * 16;          // A k-base within 32-chunk
  const int bkb  = hs * 16;          // B k-base within 32-chunk
  v8f c = {};
  const TA* arow = A + (size_t)(m0 + lr) * lda;
  for (int k = 0; k < K; k += 32) {
    v16h a, b;
#pragma unroll
    for (int t = 0; t < 8; ++t) {
      a[t]     = (_Float16)arow[k + akb + t];
      a[t + 8] = (_Float16)arow[k + akb + 8 + t];
    }
    const _Float16* bp = B + (size_t)(k + bkb) * ldb + (n0 + lr);
#pragma unroll
    for (int t = 0; t < 16; ++t) { b[t] = *bp; bp += ldb; }
    c = wmma_step(a, b, c);
  }
  return c;
}

// ------------------------------- pre-passes --------------------------------

__global__ void k_cvt16(_Float16* dst, const float* src, int n) {
  int i = blockIdx.x * blockDim.x + threadIdx.x;
  if (i < n) dst[i] = (_Float16)src[i];
}

// compact index -> padded index (g=9 for conv 3x3 tail, g=1 otherwise)
__device__ __forceinline__ int padmap(int k, int g) {
  int q = k / g, r = k - q * g;
  return ((q >> 6) * 128 + (q & 63)) * g + r;
}

// gather encoder weight: B[k][h] (K x 32, f16) from src[h][padmap(k)]
__global__ void k_enc_w(_Float16* B, const float* src, int src_ld, int K, int g) {
  int i = blockIdx.x * blockDim.x + threadIdx.x;
  if (i >= K * 32) return;
  int k = i >> 5, h = i & 31;
  B[i] = (_Float16)src[(size_t)h * src_ld + padmap(k, g)];
}

// gather decoder weight: B[h][p] (96 x Nc, f16) from src[padmap(p)][h]; bias too
__global__ void k_dec_w(_Float16* B, float* biasg, const float* src,
                        const float* bsrc, int Nc, int g) {
  int i = blockIdx.x * blockDim.x + threadIdx.x;
  if (i >= Nc * 96) return;
  int h = i / Nc, p = i - h * Nc;
  int pp = padmap(p, g);
  B[i] = (_Float16)src[(size_t)pp * 96 + h];
  if (h == 0) biasg[p] = bsrc[pp];
}

// feats: row 0 = ones; rows 1..1024 cols 64..95 = embed_tab[node_prims];
// pad rows = 0. Cols 0..63 of rows 1..1024 are written by the encoder GEMMs.
__global__ void k_feats_init(_Float16* feats, const float* embed_tab,
                             const int* node_prims) {
  int i = blockIdx.x * blockDim.x + threadIdx.x;
  if (i >= NPAD * FD) return;
  int r = i / FD, c = i - r * FD;
  float v;
  if (r == 0) v = 1.0f;
  else if (r < NN) {
    if (c < 64) return;                       // filled by encoder GEMMs
    v = embed_tab[node_prims[r - 1] * HID + (c - 64)];
  } else v = 0.0f;
  feats[i] = (_Float16)v;
}

// dinv[i] = 1/sqrt(1 + sum_j adj[i][j]); 0 on pad rows
__global__ void k_dinv(float* dinv, const int* adj) {
  int i = blockIdx.x * blockDim.x + threadIdx.x;
  if (i >= NPAD) return;
  if (i < NN) {
    int s = 1;
    const int* row = adj + (size_t)i * NN;
    for (int j = 0; j < NN; ++j) s += row[j];
    dinv[i] = rsqrtf((float)s);
  } else dinv[i] = 0.0f;
}

// Ahat = adj + I, zero-padded to NPAD x NPAD, f16 (entries in {0,1,2}: exact)
__global__ void k_ahat(_Float16* Ahat, const int* adj) {
  int i = blockIdx.x * blockDim.x + threadIdx.x;
  if (i >= NPAD * NPAD) return;
  int r = i / NPAD, c = i - r * NPAD;
  float v = 0.0f;
  if (r < NN && c < NN)
    v = (float)adj[(size_t)r * NN + c] + (r == c ? 1.0f : 0.0f);
  Ahat[i] = (_Float16)v;
}

// ------------------------------- GEMM stages -------------------------------

// enc = A(f32, MxK) @ B(f16, Kx32) + bias; write f16 into feats cols 0..31
// and duplicate into cols 32..63 (feats = [enc, enc, emb]).
__global__ void k_gemm_enc(const float* A, int lda, const _Float16* B,
                           const float* bias, int K, _Float16* feats,
                           int row_off) {
  int n0 = blockIdx.x * 16, m0 = blockIdx.y * 16;
  v8f c = wmma_tile<float>(A, lda, B, 32, K, m0, n0);
  int lane = threadIdx.x & 31, hs = lane >> 4, lr = lane & 15;
  int col = n0 + lr;
  float bv = bias[col];
#pragma unroll
  for (int v = 0; v < 8; ++v) {
    int row = 1 + row_off + m0 + hs * 8 + v;
    _Float16 val = (_Float16)(c[v] + bv);
    feats[row * FD + col]      = val;
    feats[row * FD + 32 + col] = val;
  }
}

// u = dinv[:,None] * (A @ W)   (A: NPADx96 f16, W: 96x96 f16)
__global__ void k_gemm_gcn(const _Float16* A, const _Float16* Bg,
                           const float* dinv, _Float16* u) {
  int n0 = blockIdx.x * 16, m0 = blockIdx.y * 16;
  v8f c = wmma_tile<_Float16>(A, FD, Bg, FD, FD, m0, n0);
  int lane = threadIdx.x & 31, hs = lane >> 4, lr = lane & 15;
  int col = n0 + lr;
#pragma unroll
  for (int v = 0; v < 8; ++v) {
    int row = m0 + hs * 8 + v;
    u[row * FD + col] = (_Float16)(c[v] * dinv[row]);
  }
}

// out = [relu]( dinv[:,None] * (Ahat @ u) )
__global__ void k_gemm_prop(const _Float16* Ahat, const _Float16* u,
                            const float* dinv, _Float16* out, int do_relu) {
  int n0 = blockIdx.x * 16, m0 = blockIdx.y * 16;
  v8f c = wmma_tile<_Float16>(Ahat, NPAD, u, FD, NPAD, m0, n0);
  int lane = threadIdx.x & 31, hs = lane >> 4, lr = lane & 15;
  int col = n0 + lr;
#pragma unroll
  for (int v = 0; v < 8; ++v) {
    int row = m0 + hs * 8 + v;
    float x = c[v] * dinv[row];
    if (do_relu) x = fmaxf(x, 0.0f);
    out[row * FD + col] = (_Float16)x;
  }
}

// Decoder GEMM, TDM-staged macro-tile version.
// Block = 128 threads (4 waves) computing a 64(M) x 64(N) tile of
// out(f32) = X(f16, Mx96) @ B(f16, 96xNc) + bias. Per 32-K step, wave 0
// issues one Tensor Data Mover DMA (tensor_load_to_lds) that pulls the
// 32x64 f16 B tile out of the strided weight tensor straight into LDS
// (row-major, exactly the lds_b[32][64] layout), waits on TENSORcnt, and
// a workgroup barrier publishes it. Each wave then runs 4 back-to-back
// WMMAs (one A fragment reused over 4 N subtiles).
__global__ void __launch_bounds__(128)
k_gemm_dec_blk(const _Float16* __restrict__ X, const _Float16* __restrict__ B,
               const float* __restrict__ bias, float* __restrict__ out, int Nc) {
  __shared__ alignas(16) _Float16 lds_b[32][64];
  const int tid  = threadIdx.x;
  const int wave = tid >> 5;
  const int lane = tid & 31;
  const int hs = lane >> 4, lr = lane & 15;
  const int akb = hs * 16, bkb = hs * 16;
  const int n_base = blockIdx.x * 64;
  const int m0 = blockIdx.y * 64 + wave * 16;

  v8f acc[4] = {{}, {}, {}, {}};
  const _Float16* arow = X + (size_t)(m0 + lr) * FD;

  for (int k = 0; k < FD; k += 32) {
    __syncthreads();   // protect previous iteration's LDS reads
#if defined(__gfx1250__) && __has_builtin(__builtin_amdgcn_tensor_load_to_lds)
    if (wave == 0) {
      // Tensor DMA descriptor (ISA ch.8): 2D tile 64(dim0) x 32(dim1) of
      // 2-byte elements from tensor with dim0 stride Nc, into LDS.
      unsigned lds_off = (unsigned)(size_t)(&lds_b[0][0]);
      unsigned long long ga =
          (unsigned long long)(size_t)(B + (size_t)k * Nc + n_base);
      u32x4 g0;
      g0[0] = 1u;                                   // count=1 (valid), user mode
      g0[1] = lds_off;                              // lds_addr
      g0[2] = (unsigned)(ga & 0xFFFFFFFFu);         // global_addr[31:0]
      g0[3] = (unsigned)((ga >> 32) & 0x1FFFFFFu)   // global_addr[56:32]
              | (2u << 30);                         // type=2 ("image")
      i32x8 g1;
      g1[0] = (int)(1u << 16);                      // data_size=1 -> 2 bytes
      g1[1] = (int)(((unsigned)Nc & 0xFFFFu) << 16);          // tensor_dim0 lo
      g1[2] = (int)(((unsigned)Nc >> 16) | ((unsigned)FD << 16)); // d0 hi | dim1 lo
      g1[3] = (int)(64u << 16);                     // tensor_dim1 hi=0 | tile_dim0=64
      g1[4] = 32;                                   // tile_dim1=32, tile_dim2=0
      g1[5] = Nc;                                   // tensor_dim0_stride[31:0]
      g1[6] = 0;                                    // stride hi | dim1_stride lo
      g1[7] = 0;
      i32x4 gz4 = {0, 0, 0, 0};                     // groups 2/3 unused (2D)
      i32x8 gz8 = {0, 0, 0, 0, 0, 0, 0, 0};
      __builtin_amdgcn_tensor_load_to_lds(g0, g1, gz4, gz4, gz8, 0);
      __builtin_amdgcn_s_wait_tensorcnt(0);
    }
#else
    {
      // fallback: cooperative staging, 256 16-byte chunks, 2 per thread
      int c0 = tid * 2;
#pragma unroll
      for (int c = c0; c < c0 + 2; ++c) {
        int row = c >> 3, col = (c & 7) * 8;
        const _Float16* gp = B + (size_t)(k + row) * Nc + n_base + col;
        *(v8h*)&lds_b[row][col] = *(const v8h*)gp;
      }
    }
#endif
    __syncthreads();

    v16h a;
#pragma unroll
    for (int t = 0; t < 8; ++t) {
      a[t]     = arow[k + akb + t];
      a[t + 8] = arow[k + akb + 8 + t];
    }
#pragma unroll
    for (int t = 0; t < 4; ++t) {
      v16h b;
#pragma unroll
      for (int j = 0; j < 16; ++j) b[j] = lds_b[bkb + j][t * 16 + lr];
      acc[t] = wmma_step(a, b, acc[t]);
    }
  }

#pragma unroll
  for (int t = 0; t < 4; ++t) {
    int col = n_base + t * 16 + lr;
    float bv = bias[col];
#pragma unroll
    for (int v = 0; v < 8; ++v) {
      int row = m0 + hs * 8 + v;
      out[(size_t)row * Nc + col] = acc[t][v] + bv;
    }
  }
}

// ------------------------------- launcher ----------------------------------

extern "C" void kernel_launch(void* const* d_in, const int* in_sizes, int n_in,
                              void* d_out, int out_size, void* d_ws, size_t ws_size,
                              hipStream_t stream) {
  (void)in_sizes; (void)n_in; (void)out_size; (void)ws_size;
  const float* conv_w     = (const float*)d_in[0];
  const float* lin_w      = (const float*)d_in[1];
  const float* bias_w     = (const float*)d_in[2];
  const float* conv_enc_w = (const float*)d_in[3];
  const float* conv_enc_b = (const float*)d_in[4];
  const float* lin_enc_w  = (const float*)d_in[5];
  const float* lin_enc_b  = (const float*)d_in[6];
  const float* bias_enc_w = (const float*)d_in[7];
  const float* bias_enc_b = (const float*)d_in[8];
  const float* gcn_w1     = (const float*)d_in[9];
  const float* gcn_w2     = (const float*)d_in[10];
  const float* conv_dec_w = (const float*)d_in[11];
  const float* conv_dec_b = (const float*)d_in[12];
  const float* lin_dec_w  = (const float*)d_in[13];
  const float* lin_dec_b  = (const float*)d_in[14];
  const float* bias_dec_w = (const float*)d_in[15];
  const float* bias_dec_b = (const float*)d_in[16];
  const float* embed_tab  = (const float*)d_in[17];
  const int*   adj        = (const int*)d_in[18];
  const int*   node_prims = (const int*)d_in[19];
  float* out = (float*)d_out;

  // workspace carving (256B aligned chunks)
  char* ws = (char*)d_ws;
  size_t off = 0;
  auto alloc = [&](size_t bytes) -> void* {
    void* p = ws + off;
    off += (bytes + 255) & ~(size_t)255;
    return p;
  };
  _Float16* Bce   = (_Float16*)alloc((size_t)KC * 32 * 2);
  _Float16* Ble   = (_Float16*)alloc((size_t)KL * 32 * 2);
  _Float16* Bbe   = (_Float16*)alloc((size_t)KB * 32 * 2);
  _Float16* Bg1   = (_Float16*)alloc((size_t)FD * FD * 2);
  _Float16* Bg2   = (_Float16*)alloc((size_t)FD * FD * 2);
  _Float16* feats = (_Float16*)alloc((size_t)NPAD * FD * 2);
  _Float16* ubuf  = (_Float16*)alloc((size_t)NPAD * FD * 2);
  _Float16* hbuf  = (_Float16*)alloc((size_t)NPAD * FD * 2);
  _Float16* xbuf  = (_Float16*)alloc((size_t)NPAD * FD * 2);
  _Float16* Ahat  = (_Float16*)alloc((size_t)NPAD * NPAD * 2);
  float*    dinv  = (float*)alloc((size_t)NPAD * 4);
  _Float16* Bdc   = (_Float16*)alloc((size_t)FD * KC * 2);
  float*    biasc = (float*)alloc((size_t)KC * 4);
  _Float16* Bdl   = (_Float16*)alloc((size_t)FD * KL * 2);
  float*    biasl = (float*)alloc((size_t)KL * 4);
  _Float16* Bdb   = (_Float16*)alloc((size_t)FD * KB * 2);
  float*    biasb = (float*)alloc((size_t)KB * 4);

  auto g1 = [](long long n) { return dim3((unsigned)((n + 255) / 256)); };

  // pre-passes: weight gathers / conversions / graph prep
  k_enc_w<<<g1((long long)KC * 32), 256, 0, stream>>>(Bce, conv_enc_w, PC, KC, 9);
  k_enc_w<<<g1((long long)KL * 32), 256, 0, stream>>>(Ble, lin_enc_w,  PL, KL, 1);
  k_enc_w<<<g1((long long)KB * 32), 256, 0, stream>>>(Bbe, bias_enc_w, PB, KB, 1);
  k_cvt16<<<g1(FD * FD), 256, 0, stream>>>(Bg1, gcn_w1, FD * FD);
  k_cvt16<<<g1(FD * FD), 256, 0, stream>>>(Bg2, gcn_w2, FD * FD);
  k_dec_w<<<g1((long long)KC * FD), 256, 0, stream>>>(Bdc, biasc, conv_dec_w, conv_dec_b, KC, 9);
  k_dec_w<<<g1((long long)KL * FD), 256, 0, stream>>>(Bdl, biasl, lin_dec_w,  lin_dec_b,  KL, 1);
  k_dec_w<<<g1((long long)KB * FD), 256, 0, stream>>>(Bdb, biasb, bias_dec_w, bias_dec_b, KB, 1);
  k_feats_init<<<g1(NPAD * FD), 256, 0, stream>>>(feats, embed_tab, node_prims);
  k_dinv<<<g1(NPAD), 256, 0, stream>>>(dinv, adj);
  k_ahat<<<g1((long long)NPAD * NPAD), 256, 0, stream>>>(Ahat, adj);

  // encoders -> feats cols 0..63 (duplicated)
  k_gemm_enc<<<dim3(2, NC / 16), 32, 0, stream>>>(conv_w, KC, Bce, conv_enc_b, KC, feats, 0);
  k_gemm_enc<<<dim3(2, NL / 16), 32, 0, stream>>>(lin_w,  KL, Ble, lin_enc_b,  KL, feats, NC);
  k_gemm_enc<<<dim3(2, NB / 16), 32, 0, stream>>>(bias_w, KB, Bbe, bias_enc_b, KB, feats, NC + NL);

  // GCN layer 1: h = relu(dinv * Ahat @ (dinv * (feats @ W1)))
  k_gemm_gcn <<<dim3(FD / 16, NPAD / 16), 32, 0, stream>>>(feats, Bg1, dinv, ubuf);
  k_gemm_prop<<<dim3(FD / 16, NPAD / 16), 32, 0, stream>>>(Ahat, ubuf, dinv, hbuf, 1);
  // GCN layer 2: x = dinv * Ahat @ (dinv * (h @ W2))
  k_gemm_gcn <<<dim3(FD / 16, NPAD / 16), 32, 0, stream>>>(hbuf, Bg2, dinv, ubuf);
  k_gemm_prop<<<dim3(FD / 16, NPAD / 16), 32, 0, stream>>>(Ahat, ubuf, dinv, xbuf, 0);

  // decoders -> d_out (f32), TDM-staged 64x64 macro-tiles
  k_gemm_dec_blk<<<dim3(KC / 64, NC / 64), 128, 0, stream>>>(
      xbuf + (size_t)1 * FD, Bdc, biasc, out, KC);
  k_gemm_dec_blk<<<dim3(KL / 64, NL / 64), 128, 0, stream>>>(
      xbuf + (size_t)(1 + NC) * FD, Bdl, biasl, out + (size_t)NC * KC, KL);
  k_gemm_dec_blk<<<dim3(KB / 64, NB / 64), 128, 0, stream>>>(
      xbuf + (size_t)(1 + NC + NL) * FD, Bdb, biasb,
      out + (size_t)NC * KC + (size_t)NL * KL, KB);
}
